// NoMlpGpt2Attention_33621003993400
// MI455X (gfx1250) — matrix operand order repaired and checked by reading, not compiled, not run on hardware
//
#include <hip/hip_runtime.h>
#include <hip/hip_bf16.h>

typedef __attribute__((ext_vector_type(16))) _Float16 v16h;
typedef __attribute__((ext_vector_type(8)))  float    v8f;
typedef _Float16 f16;

// Problem constants (match reference)
#define BB 2
#define SS 2048
#define EE 1024
#define HH 16
#define DD 64
#define VV 128
#define MROWS (BB*SS)          // 4096
#define HD (HH*DD)             // 1024
#define HV (HH*VV)             // 2048

#define USE_ASYNC_STAGE 1      // gfx1250 global_load_async_to_lds_b128 for K tiles

static __device__ __forceinline__ v8f wmma_f16(v16h a, v16h b, v8f c) {
    return __builtin_amdgcn_wmma_f32_16x16x32_f16(false, a, false, b, (short)0, c, false, false);
}

// ---------------------------------------------------------------------------
// Tiled GEMM: C[M,N] = epilogue(A[M,K] @ W0[K,N] [, A @ W1]).
// Block: 256 threads = 8 waves; block tile 64(M) x 128(N); K stepped by 32.
// Wave layout: 4 M-tile rows x 2 N-halves; each wave = 4 WMMA tiles (8 gated).
// B staged TRANSPOSED in LDS so B fragments are 2x ds_load_b128 each.
//  - gated (W1 != null): out = relu(acc1) * acc0
//  - else: out = (acc0 + bias) * scale  [scale /= (layer_idx+1) if ptr given]
// ---------------------------------------------------------------------------
template <typename AT>
__global__ __launch_bounds__(256)
void gemm_wmma(const AT* __restrict__ A, const float* __restrict__ W0,
               const float* __restrict__ W1, const float* __restrict__ bias,
               const int* __restrict__ layerIdxPtr, float scale,
               f16* __restrict__ outH, float* __restrict__ outF,
               int M, int K, int N) {
    __shared__ f16 As[64][40];       // [m][k]   row stride 80B = 20 banks, 16B aligned
    __shared__ f16 BsT0[128][40];    // [n][k]   transposed
    __shared__ f16 BsT1[128][40];

    const int tid  = threadIdx.x;
    const int lane = tid & 31;
    const int wv   = tid >> 5;
    const int half = lane >> 4;
    const int ln   = lane & 15;
    const int wm   = wv & 3;         // wave M-tile row (0..3)
    const int wc   = wv >> 2;        // wave N-half (0..1), 64 cols each

    const int rowBase = blockIdx.y * 64;
    const int colBase = blockIdx.x * 128;
    const bool gated  = (W1 != nullptr);

    v8f acc0[4], acc1[4];
    #pragma unroll
    for (int nt = 0; nt < 4; ++nt) { acc0[nt] = (v8f){}; acc1[nt] = (v8f){}; }

    const int ksteps = K >> 5;
    for (int kt = 0; kt < ksteps; ++kt) {
        // stage A tile 64x32 (2048 elems, 8/thread)
        #pragma unroll
        for (int it = 0; it < 8; ++it) {
            int idx = tid + it * 256;
            int r = idx >> 5, c = idx & 31;
            As[r][c] = (f16)A[(size_t)(rowBase + r) * K + kt * 32 + c];
        }
        // stage B tiles 32x128 transposed (4096 elems, 16/thread)
        #pragma unroll
        for (int it = 0; it < 16; ++it) {
            int idx = tid + it * 256;
            int r = idx >> 7, c = idx & 127;
            BsT0[c][r] = (f16)W0[(size_t)(kt * 32 + r) * N + colBase + c];
        }
        if (gated) {
            #pragma unroll
            for (int it = 0; it < 16; ++it) {
                int idx = tid + it * 256;
                int r = idx >> 7, c = idx & 127;
                BsT1[c][r] = (f16)W1[(size_t)(kt * 32 + r) * N + colBase + c];
            }
        }
        __syncthreads();

        // A fragment: M = ln, K(e) = (e&7) + (e>>3)*16 + 8*half  -> 2x b128
        v16h a;
        #pragma unroll
        for (int e = 0; e < 16; ++e) {
            int kk = (e & 7) + ((e >> 3) << 4) + (half << 3);
            a[e] = As[wm * 16 + ln][kk];
        }
        // B fragments from transposed stage: K(e) = (e&7) + (e>>3)*16 + 8*half -> 2x b128
        #pragma unroll
        for (int nt = 0; nt < 4; ++nt) {
            v16h b0;
            #pragma unroll
            for (int e = 0; e < 16; ++e)
                b0[e] = BsT0[wc * 64 + nt * 16 + ln][(e & 7) + ((e >> 3) << 4) + (half << 3)];
            acc0[nt] = wmma_f16(a, b0, acc0[nt]);
            if (gated) {
                v16h b1;
                #pragma unroll
                for (int e = 0; e < 16; ++e)
                    b1[e] = BsT1[wc * 64 + nt * 16 + ln][(e & 7) + ((e >> 3) << 4) + (half << 3)];
                acc1[nt] = wmma_f16(a, b1, acc1[nt]);
            }
        }
        __syncthreads();
    }

    float sc = scale;
    if (layerIdxPtr) sc /= (float)(*layerIdxPtr + 1);

    // C layout: M = i + 8*half, N = ln
    #pragma unroll
    for (int nt = 0; nt < 4; ++nt) {
        #pragma unroll
        for (int i = 0; i < 8; ++i) {
            int r = rowBase + wm * 16 + i + (half << 3);
            int c = colBase + wc * 64 + nt * 16 + ln;
            float val;
            if (gated) {
                val = fmaxf(acc1[nt][i], 0.0f) * acc0[nt][i];
            } else {
                val = acc0[nt][i];
                if (bias) val += bias[c];
                val *= sc;
            }
            if (outF) outF[(size_t)r * N + c] = val;
            else      outH[(size_t)r * N + c] = (f16)val;
        }
    }
}

// ---------------------------------------------------------------------------
// Flash attention: grid = B*H*(S/128), block = 256 threads (8 waves).
// Each wave: one 16-row query tile, full V=128 output (8 v8f accumulators),
// online softmax, key steps of 32. Causal mask applied analytically.
// K tile staged with gfx1250 async-to-LDS; V tile staged transposed.
// ---------------------------------------------------------------------------
__global__ __launch_bounds__(256)
void attn_wmma(const f16* __restrict__ q, const f16* __restrict__ k,
               const f16* __restrict__ v, f16* __restrict__ o) {
    __shared__ f16 Ks[32][72];       // [key_local][d], rows 144B (36 banks, 16B aligned)
    __shared__ f16 VsT[128][40];     // [vcol][key_local], transposed
    __shared__ f16 Ps[8][16][40];    // per-wave P staging

    const int tid  = threadIdx.x;
    const int lane = tid & 31;
    const int wv   = tid >> 5;
    const int half = lane >> 4;
    const int ln   = lane & 15;

    const int qchunk = blockIdx.x & 15;       // S/128 = 16
    const int bh     = blockIdx.x >> 4;
    const int b      = bh >> 4;               // H = 16
    const int h      = bh & 15;
    const int qbase  = qchunk * 128 + wv * 16;

    // preload Q fragments (A-matrix 16x64 -> 2 fragments of K=32)
    const f16* qp = q + ((size_t)(b * SS + qbase + ln) * HH + h) * DD;
    v16h qa0, qa1;
    #pragma unroll
    for (int e = 0; e < 16; ++e) {
        int d = (e & 7) + ((e >> 3) << 4) + (half << 3);
        qa0[e] = qp[d];
        qa1[e] = qp[32 + d];
    }

    v8f oacc[8];
    float mrow[8], lrow[8];
    #pragma unroll
    for (int i = 0; i < 8; ++i) { oacc[i] = (v8f){}; mrow[i] = -1e30f; lrow[i] = 0.0f; }

    // async K staging: each thread owns one 16B chunk (32 rows x 8 chunks)
    const int akr = tid >> 3;        // key row 0..31
    const int ach = tid & 7;         // 16B chunk 0..7 within the 64-f16 row

    const int nkt = (qchunk + 1) * 4;   // 32-key steps needed by this block
    for (int kt = 0; kt < nkt; ++kt) {
        const int kbase = kt * 32;
#if USE_ASYNC_STAGE
        {
            const f16* gsrc = k + ((size_t)(b * SS + kbase + akr) * HH + h) * DD + ach * 8;
            unsigned long long ga = (unsigned long long)(const void*)gsrc;
            unsigned int ldst = (unsigned int)(unsigned long long)(const void*)&Ks[akr][ach * 8];
            asm volatile("global_load_async_to_lds_b128 %0, %1, off"
                         :: "v"(ldst), "v"(ga) : "memory");
        }
#else
        #pragma unroll
        for (int it = 0; it < 8; ++it) {
            int idx = tid + it * 256;
            int kr = idx >> 6, d = idx & 63;
            Ks[kr][d] = k[((size_t)(b * SS + kbase + kr) * HH + h) * DD + d];
        }
#endif
        // stage V tile 32x128 transposed (4096 elems, 16/thread)
        #pragma unroll
        for (int it = 0; it < 16; ++it) {
            int idx = tid + it * 256;
            int kr = idx >> 7, c = idx & 127;
            VsT[c][kr] = v[((size_t)(b * SS + kbase + kr) * HH + h) * VV + c];
        }
#if USE_ASYNC_STAGE
        asm volatile("s_wait_asynccnt 0x0" ::: "memory");
#endif
        __syncthreads();

        if (kbase <= qbase + 15) {   // causal: this wave has unmasked work
            // scores: two 16x16 tiles (keys kbase..+15 and kbase+16..+31)
            v8f s0 = {}, s1 = {};
            #pragma unroll
            for (int dc = 0; dc < 2; ++dc) {
                v16h bf0, bf1;
                #pragma unroll
                for (int e = 0; e < 16; ++e) {
                    int d = dc * 32 + (e & 7) + ((e >> 3) << 4) + (half << 3);
                    bf0[e] = Ks[ln][d];        // K^T: B[d][n], n = key_local
                    bf1[e] = Ks[16 + ln][d];
                }
                s0 = wmma_f16(dc ? qa1 : qa0, bf0, s0);
                s1 = wmma_f16(dc ? qa1 : qa0, bf1, s1);
            }

            // online softmax update, per output row i (M = i + 8*half)
            #pragma unroll
            for (int i = 0; i < 8; ++i) {
                int qr = qbase + i + (half << 3);
                float x0 = s0[i]; if (kbase + ln > qr)      x0 = -1e30f;
                float x1 = s1[i]; if (kbase + 16 + ln > qr) x1 = -1e30f;
                float mx = fmaxf(x0, x1);
                #pragma unroll
                for (int off = 1; off < 16; off <<= 1)
                    mx = fmaxf(mx, __shfl_xor(mx, off, 32));
                float mnew  = fmaxf(mrow[i], mx);
                float alpha = __expf(mrow[i] - mnew);
                float p0 = (x0 < -1e29f) ? 0.0f : __expf(x0 - mnew);
                float p1 = (x1 < -1e29f) ? 0.0f : __expf(x1 - mnew);
                float lsum = p0 + p1;
                #pragma unroll
                for (int off = 1; off < 16; off <<= 1)
                    lsum += __shfl_xor(lsum, off, 32);
                lrow[i] = lrow[i] * alpha + lsum;
                mrow[i] = mnew;
                #pragma unroll
                for (int nt = 0; nt < 8; ++nt) oacc[nt][i] *= alpha;
                Ps[wv][i + (half << 3)][ln]      = (f16)p0;
                Ps[wv][i + (half << 3)][16 + ln] = (f16)p1;
            }
            // per-wave LDS round-trip: make P an A-fragment (cross-lane)
            asm volatile("s_wait_dscnt 0x0" ::: "memory");
            v16h pa;
            #pragma unroll
            for (int e = 0; e < 16; ++e) {
                int kk = (e & 7) + ((e >> 3) << 4) + (half << 3);
                pa[e] = Ps[wv][ln][kk];
            }
            // O += P @ V  (8 column tiles of V); transposed stage -> 2x b128 each
            #pragma unroll
            for (int nt = 0; nt < 8; ++nt) {
                v16h vb;
                #pragma unroll
                for (int e = 0; e < 16; ++e)
                    vb[e] = VsT[nt * 16 + ln][(e & 7) + ((e >> 3) << 4) + (half << 3)];
                oacc[nt] = wmma_f16(pa, vb, oacc[nt]);
            }
        }
        __syncthreads();
    }

    // finalize: divide by row sums, write f16 [B,S,H*V]
    #pragma unroll
    for (int i = 0; i < 8; ++i) {
        float inv = 1.0f / lrow[i];
        int qr = qbase + i + (half << 3);
        #pragma unroll
        for (int nt = 0; nt < 8; ++nt) {
            size_t idx = (size_t)(b * SS + qr) * HV + h * VV + nt * 16 + ln;
            o[idx] = (f16)(oacc[nt][i] * inv);
        }
    }
}

// ---------------------------------------------------------------------------
extern "C" void kernel_launch(void* const* d_in, const int* in_sizes, int n_in,
                              void* d_out, int out_size, void* d_ws, size_t ws_size,
                              hipStream_t stream) {
    const float* hs  = (const float*)d_in[0];
    // d_in[1] = mask (unused: causal mask applied analytically)
    const float* Wq  = (const float*)d_in[2];
    const float* bq  = (const float*)d_in[3];
    const float* Wk  = (const float*)d_in[4];
    const float* bk  = (const float*)d_in[5];
    const float* Wvf = (const float*)d_in[6];
    const float* Wvg = (const float*)d_in[7];
    const float* Wp  = (const float*)d_in[8];
    const float* bp  = (const float*)d_in[9];
    const int*   lyr = (const int*)d_in[10];
    float* out = (float*)d_out;

    // workspace layout (f16): q[4096x1024] k[4096x1024] v[4096x2048] attn[4096x2048]
    f16* qh = (f16*)d_ws;
    f16* kh = qh + (size_t)MROWS * HD;
    f16* vh = kh + (size_t)MROWS * HD;
    f16* ah = vh + (size_t)MROWS * HV;

    dim3 blk(256);
    // q = (hs @ Wq + bq) * rsqrt(D)/(layer_idx+1), stored f16
    gemm_wmma<float><<<dim3(HD / 128, MROWS / 64), blk, 0, stream>>>(
        hs, Wq, nullptr, bq, lyr, 0.125f, qh, nullptr, MROWS, EE, HD);
    // k = hs @ Wk + bk
    gemm_wmma<float><<<dim3(HD / 128, MROWS / 64), blk, 0, stream>>>(
        hs, Wk, nullptr, bk, nullptr, 1.0f, kh, nullptr, MROWS, EE, HD);
    // v = relu(hs @ Wvg) * (hs @ Wvf)
    gemm_wmma<float><<<dim3(HV / 128, MROWS / 64), blk, 0, stream>>>(
        hs, Wvf, Wvg, nullptr, nullptr, 1.0f, vh, nullptr, MROWS, EE, HV);
    // flash attention -> attn output f16 [B,S,H*V]
    attn_wmma<<<dim3(BB * HH * (SS / 128)), blk, 0, stream>>>(qh, kh, vh, ah);
    // out = attn @ Wp + bp (fp32 output)
    gemm_wmma<f16><<<dim3(EE / 128, MROWS / 64), blk, 0, stream>>>(
        ah, Wp, nullptr, bp, nullptr, 1.0f, nullptr, out, MROWS, HV, EE);
}
